// CSWinBlock_86208583565584
// MI455X (gfx1250) — matrix operand order repaired and checked by reading, not compiled, not run on hardware
//
#include <hip/hip_runtime.h>
#include <hip/hip_bf16.h>
#include <math.h>

// ---------------------------------------------------------------------------
// CSWin block for MI455X (gfx1250, wave32, WMMA).
// All GEMMs + attention einsums run on v_wmma_f32_16x16x32_f16.
// Data movement: async global->LDS b128 copies (ASYNCcnt) for straight tiles,
// 128-bit global/LDS accesses elsewhere; softmax reductions via DPP16.
// ---------------------------------------------------------------------------

typedef __attribute__((ext_vector_type(16))) _Float16 v16h;
typedef __attribute__((ext_vector_type(8)))  float    v8f;

union AFrag { v16h v; _Float16 h[16]; uint4 q[2]; };
union CFrag { v8f  v; float    f[8];  };

// Problem constants
#define MROWS 50176      // B*H*W = 16*56*56
#define CDIM  256
#define HID   1024
#define TP    416        // padded window length (392 -> 13 * 32)
#define NWH   1024       // 2 branches * 128 windows * 4 heads

// ---------------------------------------------------------------------------
// Async global->LDS copy (CDNA5 GLOBAL_LOAD_ASYNC_TO_LDS_B128, ASYNCcnt).
// Low 32 bits of a flat pointer into LDS are the DS byte offset (ISA 10.2).
// ---------------------------------------------------------------------------
__device__ inline void async_ld_b128(const void* gsrc, void* ldst) {
  uint32_t lofs = (uint32_t)(uintptr_t)ldst;
  asm volatile("global_load_async_to_lds_b128 %0, %1, off"
               :: "v"(lofs), "v"(gsrc) : "memory");
}
__device__ inline void wait_async0() {
  asm volatile("s_wait_asynccnt 0" ::: "memory");
}

// ---------------------------------------------------------------------------
// Fragment loaders (CDNA5 16-bit WMMA layouts, ISA 7.12.2), fully vectorized.
// A (16x32): lane<16: row=lane,   K = {0..7,16..23}
//            lane>=16: row=lane-16,K = {8..15,24..31}   -> two b128 loads
// ---------------------------------------------------------------------------
__device__ inline void load_a_frag(const _Float16* p, int ld, AFrag& a) {
  int lane = threadIdx.x & 31;
  const _Float16* row = p + (lane & 15) * ld;
  int base = (lane < 16) ? 0 : 8;
  a.q[0] = *(const uint4*)(row + base);
  a.q[1] = *(const uint4*)(row + base + 16);
}

// B (32x16) from a K-major (transposed) buffer: element (k, n) at p[n*ld + k].
// lane<16: col=lane, K={0..7,16..23} ; lane>=16: col=lane-16, K={8..15,24..31} -> two b128.
__device__ inline void load_bT_frag(const _Float16* p, int ld, AFrag& b) {
  int lane = threadIdx.x & 31;
  const _Float16* row = p + (lane & 15) * ld;
  int kb = (lane < 16) ? 0 : 8;
  b.q[0] = *(const uint4*)(row + kb);
  b.q[1] = *(const uint4*)(row + kb + 16);
}

__device__ inline v8f wmma_f16(const AFrag& a, const AFrag& b, const v8f& c) {
  return __builtin_amdgcn_wmma_f32_16x16x32_f16(false, a.v, false, b.v,
                                                (short)0, c, false, false);
}

// ---------------------------------------------------------------------------
// DPP16 row (16-lane) all-reduce butterflies: ror 8/4/2/1 within a DPP row.
// Lanes 0-15 and 16-31 reduce independently -- matches the C-fragment halves.
// ---------------------------------------------------------------------------
template <int CTRL>
__device__ inline float dpp_rot(float x) {
  int y = __builtin_amdgcn_update_dpp(0, __builtin_bit_cast(int, x),
                                      CTRL, 0xf, 0xf, true);
  return __builtin_bit_cast(float, y);
}
__device__ inline float row16_allmax(float x) {
  x = fmaxf(x, dpp_rot<0x128>(x));  // row_ror:8
  x = fmaxf(x, dpp_rot<0x124>(x));  // row_ror:4
  x = fmaxf(x, dpp_rot<0x122>(x));  // row_ror:2
  x = fmaxf(x, dpp_rot<0x121>(x));  // row_ror:1
  return x;
}
__device__ inline float row16_allsum(float x) {
  x += dpp_rot<0x128>(x);
  x += dpp_rot<0x124>(x);
  x += dpp_rot<0x122>(x);
  x += dpp_rot<0x121>(x);
  return x;
}

// ---------------------------------------------------------------------------
// f32 -> f16 conversion (weights, one shot)
// ---------------------------------------------------------------------------
__global__ void k_f32tof16(const float* __restrict__ in,
                           _Float16* __restrict__ out, int n) {
  int i = blockIdx.x * 256 + threadIdx.x;
  if (i < n) out[i] = (_Float16)in[i];
}

// ---------------------------------------------------------------------------
// LayerNorm over rows of 256, one row per wave, out f16
// ---------------------------------------------------------------------------
__global__ __launch_bounds__(256) void k_ln(const float* __restrict__ x,
                                            const float* __restrict__ g,
                                            const float* __restrict__ b,
                                            _Float16* __restrict__ out, int M) {
  int wid = threadIdx.x >> 5, lane = threadIdx.x & 31;
  int row = blockIdx.x * 8 + wid;
  if (row >= M) return;
  const float* xr = x + (size_t)row * CDIM;
  float v[8], s = 0.f, s2 = 0.f;
#pragma unroll
  for (int j = 0; j < 8; ++j) {
    v[j] = xr[j * 32 + lane];
    s += v[j]; s2 += v[j] * v[j];
  }
#pragma unroll
  for (int m = 1; m < 32; m <<= 1) {
    s  += __shfl_xor(s,  m, 32);
    s2 += __shfl_xor(s2, m, 32);
  }
  float mean = s * (1.f / 256.f);
  float var  = s2 * (1.f / 256.f) - mean * mean;
  float rstd = rsqrtf(var + 1e-5f);
  _Float16* orow = out + (size_t)row * CDIM;
#pragma unroll
  for (int j = 0; j < 8; ++j) {
    int c = j * 32 + lane;
    orow[c] = (_Float16)((v[j] - mean) * rstd * g[c] + b[c]);
  }
}

// ---------------------------------------------------------------------------
// WMMA GEMM: out(MxN) = A(MxK,f16) * B(KxN,f16) + bias [+res][gelu]
// Block tile 128x128, 8 waves as 4x2, wave tile 32x64 (2x4 of 16x16), BK=32.
// A tile staged with async global->LDS b128 (overlaps with the B transpose
// staging); B tile stored K-major in LDS so B fragments are two ds_load_b128.
// EPI: 0=none, 1=+res(f32), 2=exact GELU.  OF16: store f16 else f32.
// Requires M%128==0, N%128==0, K%32==0 (true for all four uses).
// ---------------------------------------------------------------------------
template <int EPI, bool OF16>
__global__ __launch_bounds__(256) void k_gemm(const _Float16* __restrict__ A,
                                              const _Float16* __restrict__ B,
                                              const float* __restrict__ bias,
                                              const float* __restrict__ res,
                                              void* __restrict__ outp,
                                              int N, int K) {
  __shared__ _Float16 As[128 * 40];  // row-major,  pitch 40 halves (16B-align)
  __shared__ _Float16 Bt[128 * 40];  // K-major: Bt[col][k], pitch 40
  int tid = threadIdx.x, lane = tid & 31, wid = tid >> 5;
  int m0 = blockIdx.y * 128, n0 = blockIdx.x * 128;
  int wm = wid >> 1, wn = wid & 1;

  CFrag acc[2][4];
#pragma unroll
  for (int i = 0; i < 2; ++i)
#pragma unroll
    for (int j = 0; j < 4; ++j)
#pragma unroll
      for (int r = 0; r < 8; ++r) acc[i][j].f[r] = 0.f;

  for (int k0 = 0; k0 < K; k0 += 32) {
    // Stage A tile (128x32): 512 async b128 copies, 2 per thread (no VGPRs)
#pragma unroll
    for (int u = tid; u < 512; u += 256) {
      int row = u >> 2, c8 = (u & 3) * 8;
      async_ld_b128(A + (size_t)(m0 + row) * K + k0 + c8, As + row * 40 + c8);
    }
    // Stage B tile (32x128) transposed into Bt[col][k]: 512 uint4, 2/thread
#pragma unroll
    for (int u = tid; u < 512; u += 256) {
      int row = u >> 4, c8 = (u & 15) * 8;  // row = k (0..31), c8 = col start
      uint4 pk = *(const uint4*)(B + (size_t)(k0 + row) * N + n0 + c8);
      const _Float16* ph = (const _Float16*)&pk;
#pragma unroll
      for (int j = 0; j < 8; ++j) Bt[(c8 + j) * 40 + row] = ph[j];
    }
    wait_async0();
    __syncthreads();

    AFrag a[2], b[4];
    load_a_frag(As + (wm * 32) * 40, 40, a[0]);
    load_a_frag(As + (wm * 32 + 16) * 40, 40, a[1]);
#pragma unroll
    for (int tj = 0; tj < 4; ++tj)
      load_bT_frag(Bt + (wn * 64 + tj * 16) * 40, 40, b[tj]);
#pragma unroll
    for (int ti = 0; ti < 2; ++ti)
#pragma unroll
      for (int tj = 0; tj < 4; ++tj)
        acc[ti][tj].v = wmma_f16(a[ti], b[tj], acc[ti][tj].v);
    __syncthreads();
  }

  // Epilogue: C layout -> row = r + 8*(lane>=16), col = lane&15
  int half8 = (lane >> 4) * 8;
#pragma unroll
  for (int ti = 0; ti < 2; ++ti)
#pragma unroll
    for (int tj = 0; tj < 4; ++tj)
#pragma unroll
      for (int r = 0; r < 8; ++r) {
        int row = m0 + wm * 32 + ti * 16 + r + half8;
        int col = n0 + wn * 64 + tj * 16 + (lane & 15);
        float v = acc[ti][tj].f[r] + bias[col];
        if (EPI == 1) v += res[(size_t)row * N + col];
        if (EPI == 2) v = 0.5f * v * (1.f + erff(v * 0.70710678118654752f));
        size_t o = (size_t)row * N + col;
        if (OF16) ((_Float16*)outp)[o] = (_Float16)v;
        else      ((float*)outp)[o] = v;
      }
}

// ---------------------------------------------------------------------------
// Scatter qkv (M x 768, f16) into padded windowed layout (pure uint4 copies):
// q/k/v : [whead(1024)][t(416)][d(32)], rows t>=392 zero-filled.
// whead = br*512 + w*4 + head ; branch0 = H-stripes (Hsp=56,Wsp=7),
// branch1 = W-stripes (Hsp=7,Wsp=56). One thread per (whead, t, d8).
// ---------------------------------------------------------------------------
__global__ __launch_bounds__(256) void k_window(const _Float16* __restrict__ qkv,
                                                _Float16* __restrict__ qw,
                                                _Float16* __restrict__ kw,
                                                _Float16* __restrict__ vw) {
  int gid = blockIdx.x * 256 + threadIdx.x;
  if (gid >= NWH * TP * 4) return;
  int d8 = (gid & 3) * 8;
  int t  = (gid >> 2) % TP;
  int whead = gid / (TP * 4);
  size_t dst = ((size_t)whead * TP + t) * 32 + d8;

  if (t >= 392) {
    uint4 z = {0u, 0u, 0u, 0u};
    *(uint4*)(qw + dst) = z;
    *(uint4*)(kw + dst) = z;
    *(uint4*)(vw + dst) = z;
    return;
  }
  int br = whead >> 9, w = (whead >> 2) & 127, head = whead & 3;
  int b = w >> 3, wi = w & 7;
  int hw;
  if (br == 0) { int r = t / 7,  cc = t % 7;  hw = r * 56 + wi * 7 + cc; }
  else         { int r = t / 56, cc = t % 56; hw = (wi * 7 + r) * 56 + cc; }
  int c = br * 128 + head * 32 + d8;
  size_t row = ((size_t)b * 3136 + hw) * 768;
  *(uint4*)(qw + dst) = *(const uint4*)(qkv + row + c);
  *(uint4*)(kw + dst) = *(const uint4*)(qkv + row + 256 + c);
  *(uint4*)(vw + dst) = *(const uint4*)(qkv + row + 512 + c);
}

// ---------------------------------------------------------------------------
// Flash attention: one block per window-head (grid=1024), 8 waves loop over
// the 26 query tiles. K staged with async global->LDS b128; V staged
// TRANSPOSED in LDS once per block; all fragments are b128 LDS loads;
// softmax reductions use DPP16. LePE 3x3 depthwise conv from the LDS V^T.
// ---------------------------------------------------------------------------
__global__ __launch_bounds__(256) void k_attn(const _Float16* __restrict__ qw,
                                              const _Float16* __restrict__ kw,
                                              const _Float16* __restrict__ vw,
                                              const float* __restrict__ cw0,
                                              const float* __restrict__ cb0,
                                              const float* __restrict__ cw1,
                                              const float* __restrict__ cb1,
                                              _Float16* __restrict__ att) {
  __shared__ _Float16 Ks[TP * 32];        // 26,624 B  row-major [key][d]
  __shared__ _Float16 Vt[32 * TP];        // 26,624 B  transposed [d][key]
  __shared__ _Float16 pbuf[8][16 * 40];   // 10,240 B  per-wave P strips

  int tid = threadIdx.x, lane = tid & 31, wid = tid >> 5;
  int whead = blockIdx.x;
  int br = whead >> 9, w = (whead >> 2) & 127, head = whead & 3;
  size_t base = (size_t)whead * (TP * 32);

  // Stage K via async copies; stage V with a register transpose (overlapped)
  for (int u = tid; u < TP * 32 / 8; u += 256)
    async_ld_b128(kw + base + (size_t)u * 8, Ks + u * 8);
  for (int u = tid; u < TP * 32 / 8; u += 256) {
    int t = u >> 2, d8 = (u & 3) * 8;
    uint4 pk = *(const uint4*)(vw + base + (size_t)t * 32 + d8);
    const _Float16* ph = (const _Float16*)&pk;
#pragma unroll
    for (int j = 0; j < 8; ++j) Vt[(d8 + j) * TP + t] = ph[j];
  }
  wait_async0();
  __syncthreads();

  int c = lane & 15;
  int half8 = (lane >> 4) * 8;
  int b_ = w >> 3, wi = w & 7;
  const float* cw = br ? cw1 : cw0;
  const float* cb = br ? cb1 : cb0;
  int Wsp = br ? 56 : 7;
  int Hsp = br ? 7 : 56;

  // Hoist LePE weights: each lane owns channels d = c and c+16
  float wk[2][9], bwt[2];
#pragma unroll
  for (int g = 0; g < 2; ++g) {
    int cl = head * 32 + c + g * 16;
    bwt[g] = cb[cl];
#pragma unroll
    for (int j = 0; j < 9; ++j) wk[g][j] = cw[cl * 9 + j];
  }

  for (int qi = 0; qi < 4; ++qi) {
    int qtile = wid + qi * 8;     // 0..31 over 8 waves
    if (qtile >= 26) break;       // wave-uniform

    AFrag aq;
    load_a_frag(qw + base + (size_t)qtile * 16 * 32, 32, aq);
#pragma unroll
    for (int e = 0; e < 16; ++e)
      aq.h[e] = aq.h[e] * (_Float16)0.17677669529663687f;  // 1/sqrt(32)

    float m[8], l[8];
    CFrag acc0, acc1;
#pragma unroll
    for (int r = 0; r < 8; ++r) {
      m[r] = -1e30f; l[r] = 0.f; acc0.f[r] = 0.f; acc1.f[r] = 0.f;
    }

    for (int kt = 0; kt < 13; ++kt) {
      int key0 = kt * 32;
      AFrag bk0, bk1;
      load_bT_frag(Ks + (size_t)key0 * 32, 32, bk0);        // keys, d-major
      load_bT_frag(Ks + (size_t)(key0 + 16) * 32, 32, bk1);
      CFrag s0, s1;
#pragma unroll
      for (int r = 0; r < 8; ++r) { s0.f[r] = 0.f; s1.f[r] = 0.f; }
      s0.v = wmma_f16(aq, bk0, s0.v);
      s1.v = wmma_f16(aq, bk1, s1.v);

      int kc0 = key0 + c, kc1 = key0 + 16 + c;
#pragma unroll
      for (int r = 0; r < 8; ++r) {
        float v0 = (kc0 < 392) ? s0.f[r] : -1e30f;
        float v1 = (kc1 < 392) ? s1.f[r] : -1e30f;
        float tmax = row16_allmax(fmaxf(v0, v1));
        float nm = fmaxf(m[r], tmax);
        float sc = __expf(m[r] - nm);
        float p0 = __expf(v0 - nm);
        float p1 = __expf(v1 - nm);
        float ps = row16_allsum(p0 + p1);
        l[r] = l[r] * sc + ps;
        m[r] = nm;
        acc0.f[r] *= sc;
        acc1.f[r] *= sc;
        int row = r + half8;  // C-layout row for this lane
        pbuf[wid][row * 40 + c]      = (_Float16)p0;
        pbuf[wid][row * 40 + 16 + c] = (_Float16)p1;
      }
      // P (16x32) as A fragment via wave-private LDS (DS ops in-order per wave)
      AFrag ap;
      load_a_frag(pbuf[wid], 40, ap);
      AFrag bv0, bv1;
      load_bT_frag(Vt + key0, TP, bv0);             // d cols 0..15
      load_bT_frag(Vt + 16 * TP + key0, TP, bv1);   // d cols 16..31
      acc0.v = wmma_f16(ap, bv0, acc0.v);
      acc1.v = wmma_f16(ap, bv1, acc1.v);
    }

    // Epilogue: normalize, add LePE from LDS V^T, scatter to (B,L,C) f16
#pragma unroll
    for (int r = 0; r < 8; ++r) {
      int t = qtile * 16 + r + half8;
      if (t >= 392) continue;
      int rw = t / Wsp, cwn = t % Wsp;
      int hw = br ? ((wi * 7 + rw) * 56 + cwn) : (rw * 56 + wi * 7 + cwn);
      float inv = 1.f / l[r];
#pragma unroll
      for (int g = 0; g < 2; ++g) {
        int d = c + g * 16;
        float o = (g ? acc1.f[r] : acc0.f[r]) * inv;
        float lp = bwt[g];
#pragma unroll
        for (int dr = -1; dr <= 1; ++dr) {
          int rr = rw + dr;
          if (rr < 0 || rr >= Hsp) continue;
#pragma unroll
          for (int dc = -1; dc <= 1; ++dc) {
            int cc2 = cwn + dc;
            if (cc2 < 0 || cc2 >= Wsp) continue;
            lp += wk[g][(dr + 1) * 3 + (dc + 1)] *
                  (float)Vt[(size_t)d * TP + rr * Wsp + cc2];
          }
        }
        att[((size_t)b_ * 3136 + hw) * CDIM + br * 128 + head * 32 + d] =
            (_Float16)(o + lp);
      }
    }
  }
}

// ---------------------------------------------------------------------------
// Host-side orchestration
// ---------------------------------------------------------------------------
extern "C" void kernel_launch(void* const* d_in, const int* in_sizes, int n_in,
                              void* d_out, int out_size, void* d_ws, size_t ws_size,
                              hipStream_t stream) {
  (void)in_sizes; (void)n_in; (void)out_size; (void)ws_size;
  const float* x      = (const float*)d_in[0];
  const float* n1g    = (const float*)d_in[3];
  const float* n1b    = (const float*)d_in[4];
  const float* qkv_w  = (const float*)d_in[5];
  const float* qkv_b  = (const float*)d_in[6];
  const float* cw0    = (const float*)d_in[7];
  const float* cb0    = (const float*)d_in[8];
  const float* cw1    = (const float*)d_in[9];
  const float* cb1    = (const float*)d_in[10];
  const float* proj_w = (const float*)d_in[11];
  const float* proj_b = (const float*)d_in[12];
  const float* n2g    = (const float*)d_in[13];
  const float* n2b    = (const float*)d_in[14];
  const float* fc1_w  = (const float*)d_in[15];
  const float* fc1_b  = (const float*)d_in[16];
  const float* fc2_w  = (const float*)d_in[17];
  const float* fc2_b  = (const float*)d_in[18];
  float* out = (float*)d_out;

  // Workspace layout (bytes). hid reuses the dead qkv + q/k/v-window region.
  char* ws = (char*)d_ws;
  const size_t SZ_QKVW = (size_t)CDIM * 768 * 2;
  const size_t SZ_PRJW = (size_t)CDIM * CDIM * 2;
  const size_t SZ_FC1W = (size_t)CDIM * HID * 2;
  const size_t SZ_FC2W = (size_t)HID * CDIM * 2;
  const size_t SZ_IMG  = (size_t)MROWS * CDIM * 2;   // 25.7 MB
  const size_t SZ_QKV  = (size_t)MROWS * 768 * 2;    // 77.1 MB
  const size_t SZ_WIN  = (size_t)NWH * TP * 32 * 2;  // 27.3 MB each

  size_t o = 0;
  _Float16* w_qkv16 = (_Float16*)(ws + o); o += SZ_QKVW;
  _Float16* w_prj16 = (_Float16*)(ws + o); o += SZ_PRJW;
  _Float16* w_fc116 = (_Float16*)(ws + o); o += SZ_FC1W;
  _Float16* w_fc216 = (_Float16*)(ws + o); o += SZ_FC2W;
  _Float16* img16   = (_Float16*)(ws + o); o += SZ_IMG;      // reused as h (LN2)
  _Float16* qkv16   = (_Float16*)(ws + o); size_t o_qkv = o; o += SZ_QKV;
  _Float16* qwbuf   = (_Float16*)(ws + o); o += SZ_WIN;
  _Float16* kwbuf   = (_Float16*)(ws + o); o += SZ_WIN;
  _Float16* vwbuf   = (_Float16*)(ws + o); o += SZ_WIN;
  _Float16* att16   = (_Float16*)(ws + o); o += SZ_IMG;
  float*    xmid    = (float*)(ws + o);    o += (size_t)MROWS * CDIM * 4;
  _Float16* hid16   = (_Float16*)(ws + o_qkv);  // 102.8 MB overlay, safe by then

  // 1. Convert weights to f16
  k_f32tof16<<<(CDIM * 768 + 255) / 256, 256, 0, stream>>>(qkv_w, w_qkv16, CDIM * 768);
  k_f32tof16<<<(CDIM * CDIM + 255) / 256, 256, 0, stream>>>(proj_w, w_prj16, CDIM * CDIM);
  k_f32tof16<<<(CDIM * HID + 255) / 256, 256, 0, stream>>>(fc1_w, w_fc116, CDIM * HID);
  k_f32tof16<<<(HID * CDIM + 255) / 256, 256, 0, stream>>>(fc2_w, w_fc216, HID * CDIM);

  // 2. LN1 -> img16
  k_ln<<<MROWS / 8, 256, 0, stream>>>(x, n1g, n1b, img16, MROWS);

  // 3. QKV GEMM (f16 out)
  k_gemm<0, true><<<dim3(768 / 128, MROWS / 128), 256, 0, stream>>>(
      img16, w_qkv16, qkv_b, nullptr, qkv16, 768, CDIM);

  // 4. Scatter into padded window layout (uint4 copies)
  k_window<<<(NWH * TP * 4 + 255) / 256, 256, 0, stream>>>(qkv16, qwbuf, kwbuf, vwbuf);

  // 5. Flash attention + LePE -> att16 (B,L,C); one block per window-head
  k_attn<<<NWH, 256, 0, stream>>>(qwbuf, kwbuf, vwbuf, cw0, cb0, cw1, cb1, att16);

  // 6. Proj GEMM + residual(x) -> xmid (f32)
  k_gemm<1, false><<<dim3(CDIM / 128, MROWS / 128), 256, 0, stream>>>(
      att16, w_prj16, proj_b, x, xmid, CDIM, CDIM);

  // 7. LN2 -> h (reuse img16)
  k_ln<<<MROWS / 8, 256, 0, stream>>>(xmid, n2g, n2b, img16, MROWS);

  // 8. FC1 + exact GELU -> hid16 (overlays dead qkv/window region)
  k_gemm<2, true><<<dim3(HID / 128, MROWS / 128), 256, 0, stream>>>(
      img16, w_fc116, fc1_b, nullptr, hid16, HID, CDIM);

  // 9. FC2 + residual(xmid) -> d_out (f32)
  k_gemm<1, false><<<dim3(CDIM / 128, MROWS / 128), 256, 0, stream>>>(
      hid16, w_fc216, fc2_b, xmid, out, CDIM, HID);
}
